// CascadedGroupCrossAttention1D_27513560498774
// MI455X (gfx1250) — matrix operand order repaired and checked by reading, not compiled, not run on hardware
//
#include <hip/hip_runtime.h>

typedef __attribute__((ext_vector_type(16))) _Float16 v16h;
typedef __attribute__((ext_vector_type(8)))  float    v8f;
typedef __attribute__((ext_vector_type(4)))  float    v4f;
typedef __attribute__((ext_vector_type(4)))  _Float16 v4h;
typedef __attribute__((ext_vector_type(4)))  int      v4i;

#define B_   32
#define L_   512
#define C_   512
#define H_   8
#define CIN  64          // C / H
#define O1_  96          // 2*kd + d
#define KD_  16
#define DV_  64
#define QKW  32          // q|k packed width
#define EPS_ 1e-5f

#if defined(__has_builtin)
#if __has_builtin(__builtin_amdgcn_global_load_async_to_lds_b128)
#define HAS_ASYNC_LDS 1
#endif
#endif

struct Params {
    const float *x, *y, *attn_bias;
    const float *qkv_w_x, *qkv_g_x, *qkv_b_x, *qkv_m_x, *qkv_v_x;
    const float *dw_w_x,  *dw_g_x,  *dw_b_x,  *dw_m_x,  *dw_v_x;
    const float *proj_w_x,*proj_g_x,*proj_b_x,*proj_m_x,*proj_v_x;
    const float *qkv_w_y, *qkv_g_y, *qkv_b_y, *qkv_m_y, *qkv_v_y;
    const float *dw_w_y,  *dw_g_y,  *dw_b_y,  *dw_m_y,  *dw_v_y;
    const float *proj_w_y,*proj_g_y,*proj_b_y,*proj_m_y,*proj_v_y;
    float*     out;
    _Float16*  ws;       // [ 2*B*C*L : relu concat f16 ][ 2*C*C : proj_w f16 ]
};

// CDNA5 wave32 WMMA f16 A-fragment (16x32, MxK): K index for element p.
// p=0..7 -> K = 8*half + p ; p=8..15 -> K = 16 + 8*half + (p-8)  (consecutive runs)
__device__ __forceinline__ int kmapA(int p, int half) {
    return ((p >> 3) << 4) + half * 8 + (((p >> 1) & 3) << 1) + (p & 1);
}

__device__ __forceinline__ float rmax16(float v) {
    #pragma unroll
    for (int off = 8; off >= 1; off >>= 1) v = fmaxf(v, __shfl_xor(v, off, 32));
    return v;
}
__device__ __forceinline__ float rsum16(float v) {
    #pragma unroll
    for (int off = 8; off >= 1; off >>= 1) v += __shfl_xor(v, off, 32);
    return v;
}

__device__ __forceinline__ void async_copy16(const void* g, void* l) {
#ifdef HAS_ASYNC_LDS
    __builtin_amdgcn_global_load_async_to_lds_b128(
        (__attribute__((address_space(1))) v4i*)const_cast<void*>(g),
        (__attribute__((address_space(3))) v4i*)l, 0, 0);
#else
    *(uint4*)l = *(const uint4*)g;
#endif
}
__device__ __forceinline__ void wait_async() {
#ifdef HAS_ASYNC_LDS
#if __has_builtin(__builtin_amdgcn_s_wait_asynccnt)
    __builtin_amdgcn_s_wait_asynccnt(0);
#else
    asm volatile("s_wait_asynccnt 0" ::: "memory");
#endif
#endif
}

// =====================================================================
// Kernel A: one block per batch. Full cascaded cross-attention in LDS.
// LDS (f16): feat[2][512][64] | qk[2][512][32] | v[2][64][512] = 320 KB
// =====================================================================
__global__ __launch_bounds__(256) void cascade_attn_kernel(Params P) {
    const int b    = blockIdx.x;
    const int tid  = threadIdx.x;
    const int w    = tid >> 5;
    const int lane = tid & 31;
    const int half = lane >> 4;
    const int n16  = lane & 15;

    extern __shared__ _Float16 sm[];
    _Float16* featB = sm;                          // 2 * 512*64   [l][c]
    _Float16* qkB   = sm + 2 * L_ * CIN;           // 2 * 512*32   [l][q0..15|k0..15]
    _Float16* vB    = qkB + 2 * L_ * QKW;          // 2 * 64*512   [d][m]

    const float scale = 0.25f;                     // kd^-0.5, kd=16
    const _Float16 hz = (_Float16)0.f;

    for (int head = 0; head < H_; ++head) {
        // -------- phase 1: accumulate channel chunk into feat (float4) ---
        for (int idx = tid; idx < 2 * L_ * (CIN / 4); idx += 256) {
            int c4 = (idx & 15) * 4;
            int l  = (idx >> 4) & 511;
            int pp = idx >> 13;
            const float* src = pp ? P.y : P.x;
            v4f g = *(const v4f*)(src + (size_t)b * (L_ * C_) +
                                  (size_t)l * C_ + head * CIN + c4);
            v4h* f = (v4h*)(featB + pp * (L_ * CIN) + l * CIN + c4);
            v4h hv;
            if (head == 0) {
                #pragma unroll
                for (int e = 0; e < 4; ++e) hv[e] = (_Float16)g[e];
            } else {
                v4h old = *f;
                #pragma unroll
                for (int e = 0; e < 4; ++e) hv[e] = (_Float16)((float)old[e] + g[e]);
            }
            *f = hv;
        }
        __syncthreads();

        // -------- phase 2: qkv pointwise GEMM + BN ----------------------
        for (int jj = w; jj < 384; jj += 8) {
            int pp  = jj / 192;
            int rem = jj % 192;
            int o0  = (rem >> 5) * 16;
            int l0  = (rem & 31) * 16;
            const float* Wq = (pp ? P.qkv_w_y : P.qkv_w_x) + head * (O1_ * CIN);
            const float* Gq = (pp ? P.qkv_g_y : P.qkv_g_x) + head * O1_;
            const float* Bq = (pp ? P.qkv_b_y : P.qkv_b_x) + head * O1_;
            const float* Mq = (pp ? P.qkv_m_y : P.qkv_m_x) + head * O1_;
            const float* Vq = (pp ? P.qkv_v_y : P.qkv_v_x) + head * O1_;
            const _Float16* f = featB + pp * (L_ * CIN);

            v8f acc = {};
            #pragma unroll
            for (int ks = 0; ks < 2; ++ks) {
                int k0 = ks * 32;
                v16h a, bf;
                #pragma unroll
                for (int p = 0; p < 16; ++p) {
                    a[p]  = (_Float16)Wq[(o0 + n16) * CIN + k0 + kmapA(p, half)];
                    bf[p] = f[(l0 + n16) * CIN + k0 + kmapA(p, half)];
                }
                acc = __builtin_amdgcn_wmma_f32_16x16x32_f16(
                        false, a, false, bf, (short)0, acc, false, false);
            }
            float vals[8];
            #pragma unroll
            for (int r = 0; r < 8; ++r) {
                int o = o0 + r + 8 * half;
                float s = Gq[o] * rsqrtf(Vq[o] + EPS_);
                vals[r] = acc[r] * s + (Bq[o] - Mq[o] * s);
            }
            if (o0 < QKW) {              // wave-uniform: q|k rows, contiguous
                _Float16* dst = qkB + pp * (L_ * QKW) + (l0 + n16) * QKW + o0 + 8 * half;
                #pragma unroll
                for (int r = 0; r < 8; ++r) dst[r] = (_Float16)vals[r];
            } else {                     // v rows, [d][m]
                _Float16* dst = vB + pp * (DV_ * L_) + (o0 - QKW + 8 * half) * L_ + l0 + n16;
                #pragma unroll
                for (int r = 0; r < 8; ++r) dst[r * L_] = (_Float16)vals[r];
            }
        }
        __syncthreads();

        // -------- phase 3: depthwise conv(5) on q + BN ------------------
        // q lives in qk[l][0..15]; rounds split on disjoint channels
        for (int round = 0; round < 4; ++round) {
            float tmp[16];
            #pragma unroll
            for (int j = 0; j < 16; ++j) {
                int idx = round * 4096 + j * 256 + tid;
                int pp  = idx >> 13;
                int rr  = idx & 8191;
                int ch  = rr >> 9;
                int l   = rr & 511;
                const float* DW = (pp ? P.dw_w_y : P.dw_w_x) + head * (KD_ * 5) + ch * 5;
                const float* Gd = (pp ? P.dw_g_y : P.dw_g_x) + head * KD_;
                const float* Bd = (pp ? P.dw_b_y : P.dw_b_x) + head * KD_;
                const float* Md = (pp ? P.dw_m_y : P.dw_m_x) + head * KD_;
                const float* Vd = (pp ? P.dw_v_y : P.dw_v_x) + head * KD_;
                const _Float16* q = qkB + pp * (L_ * QKW);
                float s = 0.f;
                #pragma unroll
                for (int t = 0; t < 5; ++t) {
                    int ll = l + t - 2;
                    float xv = (ll >= 0 && ll < L_) ? (float)q[ll * QKW + ch] : 0.f;
                    s += xv * DW[t];
                }
                float sc = Gd[ch] * rsqrtf(Vd[ch] + EPS_);
                tmp[j] = s * sc + (Bd[ch] - Md[ch] * sc);
            }
            __syncthreads();
            #pragma unroll
            for (int j = 0; j < 16; ++j) {
                int idx = round * 4096 + j * 256 + tid;
                int pp  = idx >> 13;
                int rr  = idx & 8191;
                qkB[pp * (L_ * QKW) + (rr & 511) * QKW + (rr >> 9)] = (_Float16)tmp[j];
            }
            __syncthreads();
        }

        // -------- phase 4: flash cross-attention -------------------------
        for (int t = 0; t < 8; ++t) {
            int jj = w * 8 + t;
            int po = jj >> 5;                 // output path (0 = x-out)
            int l0 = (jj & 31) << 4;
            const _Float16* qo = qkB + (1 - po) * (L_ * QKW);   // cross query
            const _Float16* kp = qkB + po * (L_ * QKW);         // k = cols 16..31
            const _Float16* vp = vB + po * (DV_ * L_);
            _Float16* scrat    = featB + po * (L_ * CIN) + l0 * CIN; // dead rows
            const float* biasrow = P.attn_bias + head * L_;

            v16h aq;
            #pragma unroll
            for (int p = 0; p < 16; ++p) {
                int c = kmapA(p, half);
                aq[p] = (c < KD_) ? qo[(l0 + n16) * QKW + c] : hz;
            }

            v8f z = {};
            v8f oacc[4] = {z, z, z, z};
            float mrun[8], lrun[8];
            #pragma unroll
            for (int r = 0; r < 8; ++r) { mrun[r] = -3.0e38f; lrun[r] = 0.f; }

            for (int mt = 0; mt < 16; ++mt) {
                int m0 = mt * 32;
                v16h bk0, bk1;
                #pragma unroll
                for (int p = 0; p < 16; ++p) {
                    int kc = kmapA(p, half);
                    bk0[p] = (kc < KD_) ? kp[(m0 + n16) * QKW + 16 + kc]      : hz;
                    bk1[p] = (kc < KD_) ? kp[(m0 + 16 + n16) * QKW + 16 + kc] : hz;
                }
                v8f s0 = {}, s1 = {};
                s0 = __builtin_amdgcn_wmma_f32_16x16x32_f16(false, aq, false, bk0,
                                                            (short)0, s0, false, false);
                s1 = __builtin_amdgcn_wmma_f32_16x16x32_f16(false, aq, false, bk1,
                                                            (short)0, s1, false, false);
                float bias0 = biasrow[m0 + n16];
                float bias1 = biasrow[m0 + 16 + n16];
                #pragma unroll
                for (int r = 0; r < 8; ++r) {
                    float sv0 = s0[r] * scale + bias0;
                    float sv1 = s1[r] * scale + bias1;
                    float mx  = rmax16(fmaxf(sv0, sv1));
                    float mn  = fmaxf(mrun[r], mx);
                    float e0  = __expf(sv0 - mn);
                    float e1  = __expf(sv1 - mn);
                    float ef  = __expf(mrun[r] - mn);
                    lrun[r]   = lrun[r] * ef + rsum16(e0 + e1);
                    mrun[r]   = mn;
                    #pragma unroll
                    for (int nt = 0; nt < 4; ++nt) oacc[nt][r] *= ef;
                    int prow = r + 8 * half;            // P scratch [row][32]
                    scrat[prow * 32 + n16]      = (_Float16)e0;
                    scrat[prow * 32 + 16 + n16] = (_Float16)e1;
                }
                // same-wave LDS store -> cross-lane load: drain DScnt
                asm volatile("s_wait_dscnt 0" ::: "memory");
                v16h apf;
                #pragma unroll
                for (int p = 0; p < 16; ++p)
                    apf[p] = scrat[n16 * 32 + kmapA(p, half)];
                #pragma unroll
                for (int nt = 0; nt < 4; ++nt) {
                    v16h bv;
                    #pragma unroll
                    for (int p = 0; p < 16; ++p)
                        bv[p] = vp[(nt * 16 + n16) * L_ + m0 + kmapA(p, half)];
                    oacc[nt] = __builtin_amdgcn_wmma_f32_16x16x32_f16(
                                 false, apf, false, bv, (short)0, oacc[nt], false, false);
                }
            }
            // write attention output (next feat), feat[l][d]
            #pragma unroll
            for (int r = 0; r < 8; ++r) {
                float inv = 1.f / lrun[r];
                _Float16* dst = featB + po * (L_ * CIN) + (l0 + r + 8 * half) * CIN + n16;
                #pragma unroll
                for (int nt = 0; nt < 4; ++nt)
                    dst[nt * 16] = (_Float16)(oacc[nt][r] * inv);
            }
        }
        __syncthreads();

        // -------- phase 5: relu -> workspace (proj input, [ch][l]) ------
        for (int idx = tid; idx < 2 * DV_ * L_; idx += 256) {
            int pp = idx >> 15;
            int rr = idx & 32767;
            int ch = rr >> 9;
            int l  = rr & 511;
            float vv = fmaxf((float)featB[pp * (L_ * CIN) + l * CIN + ch], 0.f);
            P.ws[((size_t)(pp * B_ + b) * C_ + head * DV_ + ch) * L_ + l] = (_Float16)vv;
        }
        __syncthreads();
    }
}

// =====================================================================
// Kernel W: convert proj weights to f16 in workspace (once per call)
// =====================================================================
__global__ __launch_bounds__(256) void convw_kernel(Params P) {
    _Float16* wsW = P.ws + (size_t)2 * B_ * C_ * L_;
    for (int j = 0; j < 4; ++j) {
        int idx = (blockIdx.x * 4 + j) * 256 + threadIdx.x;     // < 2*512*512
        const float* srcw = (idx >> 18) ? P.proj_w_y : P.proj_w_x;
        wsW[idx] = (_Float16)srcw[idx & (C_ * C_ - 1)];
    }
}

// =====================================================================
// Kernel B: projection GEMM + BN + transpose to (B,L,C)
// block tile: 64 (o) x 128 (l); 8 waves = 2x4, each 32x32 (2x2 WMMA tiles)
// Tiles staged with CDNA5 async global->LDS loads when available.
// =====================================================================
__global__ __launch_bounds__(256) void proj_kernel(Params P) {
    int gid = blockIdx.x;
    const int lt = gid & 3;  gid >>= 2;   // 4 l-tiles of 128
    const int ot = gid & 7;  gid >>= 3;   // 8 o-tiles of 64
    const int b  = gid & 31; gid >>= 5;
    const int pp = gid;                   // path
    const int o0 = ot * 64, l0 = lt * 128;

    const int tid  = threadIdx.x;
    const int w    = tid >> 5;
    const int lane = tid & 31;
    const int half = lane >> 4;
    const int n16  = lane & 15;
    const int wo   = w >> 2;              // 0..1
    const int wl   = w & 3;               // 0..3

    __shared__ _Float16 Wt[64 * 32];      // [o][k]
    __shared__ _Float16 Xt[32 * 128];     // [k][l]

    const _Float16* Wsrc = P.ws + (size_t)2 * B_ * C_ * L_ + (size_t)pp * (C_ * C_);
    const _Float16* X    = P.ws + (size_t)(pp * B_ + b) * C_ * L_;

    v8f z = {};
    v8f acc[2][2] = {{z, z}, {z, z}};

    for (int kk = 0; kk < 16; ++kk) {
        int k0 = kk * 32;
        {   // W tile: 64x32 f16 = 256 chunks of 16B, one per thread
            int ro = tid >> 2, cc = tid & 3;
            async_copy16(Wsrc + (o0 + ro) * C_ + k0 + cc * 8, &Wt[ro * 32 + cc * 8]);
        }
        #pragma unroll
        for (int j = 0; j < 2; ++j) {   // X tile: 32x128 f16 = 512 chunks
            int xi = j * 256 + tid;
            int xr = xi >> 4, xc = xi & 15;
            async_copy16(X + (size_t)(k0 + xr) * L_ + l0 + xc * 8,
                         &Xt[xr * 128 + xc * 8]);
        }
        wait_async();
        __syncthreads();

        v16h a0, a1, b0, b1;
        #pragma unroll
        for (int p = 0; p < 16; ++p) {
            int ka = kmapA(p, half);
            a0[p] = Wt[(wo * 32 + n16) * 32 + ka];
            a1[p] = Wt[(wo * 32 + 16 + n16) * 32 + ka];
            b0[p] = Xt[ka * 128 + wl * 32 + n16];
            b1[p] = Xt[ka * 128 + wl * 32 + 16 + n16];
        }
        acc[0][0] = __builtin_amdgcn_wmma_f32_16x16x32_f16(false, a0, false, b0,
                                                           (short)0, acc[0][0], false, false);
        acc[0][1] = __builtin_amdgcn_wmma_f32_16x16x32_f16(false, a0, false, b1,
                                                           (short)0, acc[0][1], false, false);
        acc[1][0] = __builtin_amdgcn_wmma_f32_16x16x32_f16(false, a1, false, b0,
                                                           (short)0, acc[1][0], false, false);
        acc[1][1] = __builtin_amdgcn_wmma_f32_16x16x32_f16(false, a1, false, b1,
                                                           (short)0, acc[1][1], false, false);
        __syncthreads();
    }

    const float* G  = pp ? P.proj_g_y : P.proj_g_x;
    const float* Bb = pp ? P.proj_b_y : P.proj_b_x;
    const float* Mm = pp ? P.proj_m_y : P.proj_m_x;
    const float* Vv = pp ? P.proj_v_y : P.proj_v_x;
    float* out = P.out + (size_t)pp * (B_ * L_ * C_) + (size_t)b * (L_ * C_);

    #pragma unroll
    for (int i = 0; i < 2; ++i)
        #pragma unroll
        for (int j = 0; j < 2; ++j)
            #pragma unroll
            for (int r = 0; r < 8; ++r) {
                int o = o0 + wo * 32 + i * 16 + r + 8 * half;
                int l = l0 + wl * 32 + j * 16 + n16;
                float s = G[o] * rsqrtf(Vv[o] + EPS_);
                out[(size_t)l * C_ + o] = acc[i][j][r] * s + (Bb[o] - Mm[o] * s);
            }
}

// =====================================================================
extern "C" void kernel_launch(void* const* d_in, const int* in_sizes, int n_in,
                              void* d_out, int out_size, void* d_ws, size_t ws_size,
                              hipStream_t stream) {
    (void)in_sizes; (void)n_in; (void)out_size; (void)ws_size;
    Params P;
    P.x         = (const float*)d_in[0];
    P.y         = (const float*)d_in[1];
    P.attn_bias = (const float*)d_in[2];
    P.qkv_w_x  = (const float*)d_in[3];  P.qkv_g_x  = (const float*)d_in[4];
    P.qkv_b_x  = (const float*)d_in[5];  P.qkv_m_x  = (const float*)d_in[6];
    P.qkv_v_x  = (const float*)d_in[7];
    P.dw_w_x   = (const float*)d_in[8];  P.dw_g_x   = (const float*)d_in[9];
    P.dw_b_x   = (const float*)d_in[10]; P.dw_m_x   = (const float*)d_in[11];
    P.dw_v_x   = (const float*)d_in[12];
    P.proj_w_x = (const float*)d_in[13]; P.proj_g_x = (const float*)d_in[14];
    P.proj_b_x = (const float*)d_in[15]; P.proj_m_x = (const float*)d_in[16];
    P.proj_v_x = (const float*)d_in[17];
    P.qkv_w_y  = (const float*)d_in[18]; P.qkv_g_y  = (const float*)d_in[19];
    P.qkv_b_y  = (const float*)d_in[20]; P.qkv_m_y  = (const float*)d_in[21];
    P.qkv_v_y  = (const float*)d_in[22];
    P.dw_w_y   = (const float*)d_in[23]; P.dw_g_y   = (const float*)d_in[24];
    P.dw_b_y   = (const float*)d_in[25]; P.dw_m_y   = (const float*)d_in[26];
    P.dw_v_y   = (const float*)d_in[27];
    P.proj_w_y = (const float*)d_in[28]; P.proj_g_y = (const float*)d_in[29];
    P.proj_b_y = (const float*)d_in[30]; P.proj_m_y = (const float*)d_in[31];
    P.proj_v_y = (const float*)d_in[32];
    P.out = (float*)d_out;
    P.ws  = (_Float16*)d_ws;

    const int ldsA = 2 * (CIN + QKW + DV_) * L_ * (int)sizeof(_Float16); // 320 KB
    (void)hipFuncSetAttribute((const void*)cascade_attn_kernel,
                              hipFuncAttributeMaxDynamicSharedMemorySize, ldsA);

    convw_kernel<<<dim3(512), dim3(256), 0, stream>>>(P);
    cascade_attn_kernel<<<dim3(B_), dim3(256), ldsA, stream>>>(P);
    proj_kernel<<<dim3(2 * B_ * 8 * 4), dim3(256), 0, stream>>>(P);
}
